// MultiHeadAttention_2911987827364
// MI455X (gfx1250) — matrix operand order repaired and checked by reading, not compiled, not run on hardware
//
#include <hip/hip_runtime.h>
#include <hip/hip_bf16.h>
#include <math.h>

typedef _Float16 f16;
typedef __attribute__((ext_vector_type(16))) _Float16 v16h;
typedef __attribute__((ext_vector_type(8)))  _Float16 v8h;
typedef __attribute__((ext_vector_type(8)))  float    v8f;

#define B_  2
#define L_  2048
#define D_  1024
#define H_  16
#define DK_ 64
#define DV_ 64
#define ML_ (B_*L_)   // 4096 total rows

static __device__ __forceinline__ v8f wmma_f16(const v16h& a, const v16h& b, const v8f& c) {
    return __builtin_amdgcn_wmma_f32_16x16x32_f16(false, a, false, b, (short)0, c, false, false);
}

// A-fragment for 16x32 f16 tile: lane holds row m=lane%16; halves 0..15 = contiguous
// k-chunk at (lane/16)*16 (matches the contiguous B-fragment k order).
static __device__ __forceinline__ v16h load_a_frag(const f16* p0, const f16* p1) {
    v8h lo = *(const v8h*)p0;
    v8h hi = *(const v8h*)p1;
    v16h r;
#pragma unroll
    for (int i = 0; i < 8; ++i) { r[i] = lo[i]; r[8 + i] = hi[i]; }
    return r;
}

// CDNA5 async global->LDS copy (16B per lane), tracked by ASYNCcnt (ISA §10.7, §15.18.3).
static __device__ __forceinline__ void async_g2l_b128(f16* lds_dst, const f16* gsrc) {
    unsigned laddr = (unsigned)(unsigned long long)lds_dst;  // LDS aperture: addr[31:0]
    asm volatile("global_load_async_to_lds_b128 %0, %1, off"
                 :: "v"(laddr), "v"(gsrc) : "memory");
}
static __device__ __forceinline__ void wait_async0() {
    asm volatile("s_wait_asynccnt 0x0" ::: "memory");
}

// ---------------- conversion / packing ----------------

__global__ void cvt_f32_f16(const float* __restrict__ in, f16* __restrict__ out, int n) {
    for (int i = blockIdx.x * blockDim.x + threadIdx.x; i < n; i += gridDim.x * blockDim.x)
        out[i] = (f16)in[i];
}

// W: (H, D, 64) f32  ->  P: (H, 32, 64, 32) f16 with P[h][kb][n][kk] = W[h][kb*32+kk][n]
// so a B-fragment (lane: N=n, K=(lane/16)*16+i contiguous) is one 32-byte read.
__global__ void pack_w(const float* __restrict__ W, f16* __restrict__ P) {
    const int total = H_ * 32 * 64 * 32;
    for (int i = blockIdx.x * blockDim.x + threadIdx.x; i < total; i += gridDim.x * blockDim.x) {
        int kk = i & 31;
        int n  = (i >> 5) & 63;
        int kb = (i >> 11) & 31;
        int h  = i >> 16;
        P[i] = (f16)W[(h * D_ + kb * 32 + kk) * 64 + n];
    }
}

// ---------------- fused Q+K projection GEMM ----------------
// Block = 1 head x 8 row-tiles (8 waves). Each k-step stages the head's Q,K 64x32
// weight tiles (2 x 4KB) once per block via pipelined async copy into double-buffered
// LDS. One A fragment feeds 8 WMMAs (2 matrices x 4 n-tiles) -> 2x A reuse, and only
// 8 live accumulators (64 VGPRs) so WMMA operands stay in place (no v_mov repacking).
__global__ __launch_bounds__(256)
void qk_gemm_fused(const f16* __restrict__ Xh,
                   const f16* __restrict__ Wqp, const f16* __restrict__ Wkp,
                   const float* __restrict__ bq, const float* __restrict__ bk,
                   f16* __restrict__ Qb, f16* __restrict__ Kb) {
    __shared__ __align__(16) f16 sW[2][2][64 * 32];   // 16KB double-buffered staging
    const int t = threadIdx.x;
    const int lane = t & 31, hi = lane >> 4, lr = lane & 15;
    const int wv = t >> 5;
    const int h = blockIdx.x >> 5;                   // 16 heads
    const int rowtile = (blockIdx.x & 31) * 8 + wv;  // 256 row tiles of 16
    const int m0 = rowtile * 16;

    const f16* arow = Xh + (long)(m0 + lr) * D_;
    const long hoff = (long)h * (32 * 64 * 32);
    const f16* wq = Wqp + hoff;
    const f16* wk = Wkp + hoff;

    v8f cq[4] = {}, ck[4] = {};

    // prologue: stage tile 0
    async_g2l_b128(sW[0][0] + t * 8, wq + t * 8);
    async_g2l_b128(sW[0][1] + t * 8, wk + t * 8);

    for (int kb = 0; kb < 32; ++kb) {
        wait_async0();          // this wave's copies for tile kb are done
        __syncthreads();        // everyone's copies done; readers of kb-2 parity done
        if (kb + 1 < 32) {      // prefetch tile kb+1, overlapped with compute below
            const int nk = (kb + 1) * 2048;
            async_g2l_b128(sW[(kb + 1) & 1][0] + t * 8, wq + nk + t * 8);
            async_g2l_b128(sW[(kb + 1) & 1][1] + t * 8, wk + nk + t * 8);
        }
        const int k0 = kb * 32;
        v16h a = load_a_frag(arow + k0 + hi * 16, arow + k0 + hi * 16 + 8);
        const f16* bufq = sW[kb & 1][0];
        const f16* bufk = sW[kb & 1][1];
#pragma unroll
        for (int nt = 0; nt < 4; ++nt) {
            v16h b = *(const v16h*)(bufq + (nt * 16 + lr) * 32 + hi * 16);
            cq[nt] = wmma_f16(a, b, cq[nt]);
        }
#pragma unroll
        for (int nt = 0; nt < 4; ++nt) {
            v16h b = *(const v16h*)(bufk + (nt * 16 + lr) * 32 + hi * 16);
            ck[nt] = wmma_f16(a, b, ck[nt]);
        }
    }

    // epilogue: bias + store, natural (B,H,L,64)
#pragma unroll
    for (int nt = 0; nt < 4; ++nt) {
        const int col = nt * 16 + lr;
        const float bvq = bq[h * 64 + col];
        const float bvk = bk[h * 64 + col];
#pragma unroll
        for (int j = 0; j < 8; ++j) {
            const int m = m0 + j + 8 * hi;      // global row over B*L
            const int bb = m >> 11, lrow = m & (L_ - 1);
            const long bh = (long)(bb * H_ + h);
            Qb[(bh * L_ + lrow) * 64 + col] = (f16)(cq[nt][j] + bvq);
            Kb[(bh * L_ + lrow) * 64 + col] = (f16)(ck[nt][j] + bvk);
        }
    }
}

// ---------------- V projection GEMM (transposed store) ----------------
__global__ __launch_bounds__(256)
void v_gemm(const f16* __restrict__ Xh, const f16* __restrict__ Wvp,
            const float* __restrict__ bv, f16* __restrict__ Vt) {
    __shared__ __align__(16) f16 sW[2][64 * 32];     // 8KB double-buffered staging
    const int t = threadIdx.x;
    const int lane = t & 31, hi = lane >> 4, lr = lane & 15;
    const int wv = t >> 5;
    const int h = blockIdx.x >> 5;                   // 16 heads
    const int rowtile = (blockIdx.x & 31) * 8 + wv;  // 256 row tiles of 16
    const int m0 = rowtile * 16;

    const f16* arow = Xh + (long)(m0 + lr) * D_;
    const f16* wb = Wvp + (long)h * (32 * 64 * 32);

    v8f c[4] = {};
    async_g2l_b128(sW[0] + t * 8, wb + t * 8);
    for (int kb = 0; kb < 32; ++kb) {
        wait_async0();
        __syncthreads();
        if (kb + 1 < 32)
            async_g2l_b128(sW[(kb + 1) & 1] + t * 8, wb + (kb + 1) * 2048 + t * 8);
        const int k0 = kb * 32;
        v16h a = load_a_frag(arow + k0 + hi * 16, arow + k0 + hi * 16 + 8);
        const f16* buf = sW[kb & 1];
#pragma unroll
        for (int nt = 0; nt < 4; ++nt) {
            v16h b = *(const v16h*)(buf + (nt * 16 + lr) * 32 + hi * 16);
            c[nt] = wmma_f16(a, b, c[nt]);
        }
    }
#pragma unroll
    for (int nt = 0; nt < 4; ++nt) {
        const int col = nt * 16 + lr;
        const float bvv = bv[h * 64 + col];
#pragma unroll
        for (int j = 0; j < 8; ++j) {
            const int m = m0 + j + 8 * hi;
            const int bb = m >> 11, lrow = m & (L_ - 1);
            Vt[((long)(bb * H_ + h) * 64 + col) * L_ + lrow] = (f16)(c[nt][j] + bvv);
        }
    }
}

// ---------------- flash attention ----------------
// Block = one (b,h) x 8 query tiles (8 waves, 128 query rows). K/V tiles for each
// 32-key step are async-copied (pipelined) into double-buffered LDS and shared.
__global__ __launch_bounds__(256)
void flash_attn(const f16* __restrict__ Q, const f16* __restrict__ K,
                const f16* __restrict__ Vt, f16* __restrict__ Z) {
    __shared__ __align__(16) f16 sK[2][32 * 64];    // keys x dk   (2 x 4KB)
    __shared__ __align__(16) f16 sV[2][64 * 32];    // dv  x keys  (2 x 4KB)
    __shared__ __align__(16) f16 sP[8][16 * 32];    // per-wave P tile
    const int t = threadIdx.x;
    const int lane = t & 31, hi = lane >> 4, lr = lane & 15;
    const int wv = t >> 5;
    const int bhidx = blockIdx.x >> 4;              // 0..31 = b*16+h
    const int b = bhidx >> 4, h = bhidx & 15;
    const int qt = (blockIdx.x & 15) * 8 + wv;      // query tile 0..127
    const long bh = (long)(b * H_ + h);

    const f16* qbase = Q + (bh * L_ + qt * 16 + lr) * DK_;
    const v16h qa0 = load_a_frag(qbase + hi * 16,      qbase + 8 + hi * 16);
    const v16h qa1 = load_a_frag(qbase + 32 + hi * 16, qbase + 40 + hi * 16);

    v8f o[4] = {};
    float mstate[8], lstate[8];
#pragma unroll
    for (int j = 0; j < 8; ++j) { mstate[j] = -1e30f; lstate[j] = 0.0f; }
    const float scale = 0.125f;         // 1/sqrt(64)
    const float l2e = 1.44269504f;
    f16* pw = sP[wv];

    // per-thread staging assignments (16B each)
    const int kKey = t >> 3, kChunk = t & 7;        // K: 32 keys x 8 chunks
    const int vDv = t >> 2, vChunk = t & 3;         // V: 64 dv  x 4 chunks
    const f16* gK0 = K + (bh * L_ + kKey) * DK_ + kChunk * 8;
    const f16* gV0 = Vt + (bh * DV_ + vDv) * (long)L_ + vChunk * 8;
    const int ldsK_off = kKey * 64 + kChunk * 8;
    const int ldsV_off = vDv * 32 + vChunk * 8;

    // prologue: stage key-step 0
    async_g2l_b128(sK[0] + ldsK_off, gK0);
    async_g2l_b128(sV[0] + ldsV_off, gV0);

    for (int jb = 0; jb < L_; jb += 32) {
        const int par = (jb >> 5) & 1;
        wait_async0();
        __syncthreads();
        if (jb + 32 < L_) {     // prefetch next K/V tile, overlapped with compute
            async_g2l_b128(sK[par ^ 1] + ldsK_off, gK0 + (long)(jb + 32) * DK_);
            async_g2l_b128(sV[par ^ 1] + ldsV_off, gV0 + (jb + 32));
        }
        const f16* bufK = sK[par];
        const f16* bufV = sV[par];

        // ---- S = Q K^T for 32 keys (two 16-key n-tiles), B-frags from LDS ----
        v8f s0 = {}, s1 = {};
        {
            v16h k00 = *(const v16h*)(bufK + (0 * 16 + lr) * 64 + 0  + hi * 16);
            v16h k01 = *(const v16h*)(bufK + (0 * 16 + lr) * 64 + 32 + hi * 16);
            v16h k10 = *(const v16h*)(bufK + (1 * 16 + lr) * 64 + 0  + hi * 16);
            v16h k11 = *(const v16h*)(bufK + (1 * 16 + lr) * 64 + 32 + hi * 16);
            s0 = wmma_f16(qa0, k00, s0); s0 = wmma_f16(qa1, k01, s0);
            s1 = wmma_f16(qa0, k10, s1); s1 = wmma_f16(qa1, k11, s1);
        }

        // ---- online softmax (rows split across lane halves: row = j + 8*(lane/16)) ----
        float p0[8], p1[8], corr[8];
#pragma unroll
        for (int j = 0; j < 8; ++j) {
            float a0 = s0[j] * scale, a1 = s1[j] * scale;
            float mj = fmaxf(a0, a1);
            mj = fmaxf(mj, __shfl_xor(mj, 1, 16));
            mj = fmaxf(mj, __shfl_xor(mj, 2, 16));
            mj = fmaxf(mj, __shfl_xor(mj, 4, 16));
            mj = fmaxf(mj, __shfl_xor(mj, 8, 16));
            const float mnew = fmaxf(mstate[j], mj);
            corr[j] = exp2f((mstate[j] - mnew) * l2e);
            p0[j] = exp2f((a0 - mnew) * l2e);
            p1[j] = exp2f((a1 - mnew) * l2e);
            float ps = p0[j] + p1[j];
            ps += __shfl_xor(ps, 1, 16);
            ps += __shfl_xor(ps, 2, 16);
            ps += __shfl_xor(ps, 4, 16);
            ps += __shfl_xor(ps, 8, 16);
            lstate[j] = lstate[j] * corr[j] + ps;
            mstate[j] = mnew;
        }
#pragma unroll
        for (int nt = 0; nt < 4; ++nt)
#pragma unroll
            for (int j = 0; j < 8; ++j) o[nt][j] *= corr[j];

        // ---- C-layout -> A-fragment via per-wave LDS tile (same-wave DS ops in order) ----
#pragma unroll
        for (int j = 0; j < 8; ++j) {
            const int row = j + 8 * hi;
            pw[row * 32 + lr]      = (f16)p0[j];
            pw[row * 32 + 16 + lr] = (f16)p1[j];
        }
        v16h pa = load_a_frag(pw + lr * 32 + hi * 16, pw + lr * 32 + 8 + hi * 16);

        // ---- O += P * V : V^T B-frags from shared LDS tile ----
#pragma unroll
        for (int nt = 0; nt < 4; ++nt) {
            v16h bv = *(const v16h*)(bufV + (nt * 16 + lr) * 32 + hi * 16);
            o[nt] = wmma_f16(pa, bv, o[nt]);
        }
    }

    // ---- finalize: divide by row sums, write Z (B*L, H*DV) f16 ----
#pragma unroll
    for (int j = 0; j < 8; ++j) {
        const float rinv = 1.0f / lstate[j];
        const int m = qt * 16 + j + 8 * hi;
        f16* zrow = Z + ((long)(b * L_ + m)) * (H_ * DV_) + h * 64;
#pragma unroll
        for (int nt = 0; nt < 4; ++nt)
            zrow[nt * 16 + lr] = (f16)(o[nt][j] * rinv);
    }
}

// ---------------- output projection: out = Z @ Wo^T + bo (f32 out) ----------------
// Block = 1 col-tile(64) x 8 row-tiles; Wo tile pipelined through double-buffered LDS.
__global__ __launch_bounds__(256)
void out_proj(const f16* __restrict__ Zh, const f16* __restrict__ Woh,
              const float* __restrict__ bo, float* __restrict__ out) {
    __shared__ __align__(16) f16 sW[2][64 * 32];
    const int t = threadIdx.x;
    const int lane = t & 31, hi = lane >> 4, lr = lane & 15;
    const int wv = t >> 5;
    const int ct = blockIdx.x >> 5;                  // 16 col tiles of 64
    const int rowtile = (blockIdx.x & 31) * 8 + wv;  // 256 row tiles
    const int m0 = rowtile * 16;

    const f16* arow = Zh + (long)(m0 + lr) * D_;
    // staging: 64 rows of Wo (d = ct*64 + t/4), 32 k-halves each; 16B per thread
    const f16* gW0 = Woh + (long)(ct * 64 + (t >> 2)) * D_ + (t & 3) * 8;
    const int ldsW_off = (t >> 2) * 32 + (t & 3) * 8;

    v8f c[4] = {};
    async_g2l_b128(sW[0] + ldsW_off, gW0);
    for (int kb = 0; kb < 32; ++kb) {
        wait_async0();
        __syncthreads();
        if (kb + 1 < 32)
            async_g2l_b128(sW[(kb + 1) & 1] + ldsW_off, gW0 + (kb + 1) * 32);
        const f16* buf = sW[kb & 1];

        const int k0 = kb * 32;
        v16h a = load_a_frag(arow + k0 + hi * 16, arow + k0 + hi * 16 + 8);
#pragma unroll
        for (int nt = 0; nt < 4; ++nt) {
            // B[k][d] = Wo[d][k]: k-contiguous row of Wo = exact B fragment
            v16h b = *(const v16h*)(buf + (nt * 16 + lr) * 32 + hi * 16);
            c[nt] = wmma_f16(a, b, c[nt]);
        }
    }
#pragma unroll
    for (int nt = 0; nt < 4; ++nt) {
        const int d = ct * 64 + nt * 16 + lr;
        const float bv = bo[d];
#pragma unroll
        for (int j = 0; j < 8; ++j)
            out[(long)(m0 + j + 8 * hi) * D_ + d] = c[nt][j] + bv;
    }
}

extern "C" void kernel_launch(void* const* d_in, const int* in_sizes, int n_in,
                              void* d_out, int out_size, void* d_ws, size_t ws_size,
                              hipStream_t stream) {
    const float* X  = (const float*)d_in[0];
    const float* Wq = (const float*)d_in[1];
    const float* bq = (const float*)d_in[2];
    const float* Wk = (const float*)d_in[3];
    const float* bk = (const float*)d_in[4];
    const float* Wv = (const float*)d_in[5];
    const float* bv = (const float*)d_in[6];
    const float* Wo = (const float*)d_in[7];
    const float* bo = (const float*)d_in[8];
    float* out = (float*)d_out;

    size_t off = 0;
    char* base = (char*)d_ws;
    auto carve = [&](size_t bytes) -> void* {
        void* p = base + off;
        off += (bytes + 255) & ~(size_t)255;
        return p;
    };
    f16* Xh  = (f16*)carve((size_t)ML_ * D_ * 2);            // 8 MB
    f16* Qb  = (f16*)carve((size_t)B_ * H_ * L_ * DK_ * 2);  // 8 MB
    f16* Kb  = (f16*)carve((size_t)B_ * H_ * L_ * DK_ * 2);  // 8 MB
    f16* Vt  = (f16*)carve((size_t)B_ * H_ * DV_ * L_ * 2);  // 8 MB (transposed)
    f16* Zb  = (f16*)carve((size_t)ML_ * H_ * DV_ * 2);      // 8 MB
    f16* Wqp = (f16*)carve((size_t)H_ * D_ * 64 * 2);        // 2 MB
    f16* Wkp = (f16*)carve((size_t)H_ * D_ * 64 * 2);
    f16* Wvp = (f16*)carve((size_t)H_ * D_ * 64 * 2);
    f16* Woh = (f16*)carve((size_t)D_ * D_ * 2);             // 2 MB
    (void)ws_size; (void)in_sizes; (void)n_in; (void)out_size;

    cvt_f32_f16<<<1024, 256, 0, stream>>>(X, Xh, ML_ * D_);
    cvt_f32_f16<<<1024, 256, 0, stream>>>(Wo, Woh, D_ * D_);
    pack_w<<<512, 256, 0, stream>>>(Wq, Wqp);
    pack_w<<<512, 256, 0, stream>>>(Wk, Wkp);
    pack_w<<<512, 256, 0, stream>>>(Wv, Wvp);

    // 512 blocks: 16 heads x 32 row-groups (8 waves each)
    qk_gemm_fused<<<512, 256, 0, stream>>>(Xh, Wqp, Wkp, bq, bk, Qb, Kb);
    v_gemm<<<512, 256, 0, stream>>>(Xh, Wvp, bv, Vt);

    // 512 blocks: 32 (b,h) x 16 query-groups (8 waves each)
    flash_attn<<<512, 256, 0, stream>>>(Qb, Kb, Vt, Zb);

    // 512 blocks: 16 col-tiles x 32 row-groups
    out_proj<<<512, 256, 0, stream>>>(Zb, Woh, bo, out);
}
